// GATAutoencoder_4827543241245
// MI455X (gfx1250) — compile-verified, matrix operand order checked by reading, not run on hardware
//
#include <hip/hip_runtime.h>
#include <hip/hip_bf16.h>
#include <stdint.h>

#define NNODE 50000
#define NEDGE 800000
#define TOTE  (NNODE + NEDGE)

typedef __attribute__((ext_vector_type(16))) _Float16 v16h;
typedef __attribute__((ext_vector_type(8)))  float    v8f;

static __device__ __forceinline__ unsigned pack2h(float a, float b) {
  union { _Float16 h[2]; unsigned u; } p;
  p.h[0] = (_Float16)a; p.h[1] = (_Float16)b;
  return p.u;
}

// monotone float <-> uint key for atomic float-max via atomicMax(u32)
static __device__ __forceinline__ unsigned fkey(float f) {
  unsigned u = __float_as_uint(f);
  return (u & 0x80000000u) ? ~u : (u | 0x80000000u);
}
static __device__ __forceinline__ float funkey(unsigned k) {
  unsigned u = (k & 0x80000000u) ? (k ^ 0x80000000u) : ~k;
  return __uint_as_float(u);
}
#define NEG_INF_KEY 0x007FFFFFu  // fkey(-inf)

// ---------------------------------------------------------------------------
// WMMA GEMM: C[M,K] = A[M,F] @ W[F,K] (+ bias), fp32 in/out, f16 multiply.
// Block = 256 threads = 8 waves. Tile = 64 rows x K cols.
// Wave w: row strip (w&3)*16, column half (w>>2)*(K/2).
//
// LDS layout (packed f16 pairs as uint):
//   lA[row*AS + kp]  row-major A rows
//   lB[col*AS + kp]  TRANSPOSED W (pairs of a column contiguous)
// AS = F/2 + 4: multiple of 4 (16B-aligned uint4 quads), bank stride 4 mod 64
// -> fragment = two ds_load_b128 per matrix, conflict-free across 16 lanes.
// Fragment quads: pair indices 16g + 4*half + {0..3} and 16g + 8 + 4*half + {0..3}
// (ISA 7.12.2 16-bit layout: VGPR v -> k0 = 32g + (v>>2)*16 + half*8 + (v&3)*2).
// ---------------------------------------------------------------------------
template<int F, int K, bool BIAS>
__global__ void __launch_bounds__(256)
gemm_wmma(const float* __restrict__ A, const float* __restrict__ W,
          const float* __restrict__ bias, float* __restrict__ C, int M)
{
  constexpr int FP = F / 2;       // packed f16 pairs per column/row
  constexpr int AS = FP + 4;      // LDS stride in uints
  __shared__ unsigned lA[64 * AS];
  __shared__ unsigned lB[K * AS];

  const int tid  = threadIdx.x;
  const int row0 = blockIdx.x * 64;
  const bool fullTile = (row0 + 64 <= M);

  // stage W (F x K) transposed: lB[n*AS + kp] = {W[2kp][n], W[2kp+1][n]}
  for (int i = tid; i < FP * K; i += 256) {
    int kp = i / K, n = i % K;     // coalesced global reads over n
    lB[n * AS + kp] = pack2h(W[(2 * kp) * K + n], W[(2 * kp + 1) * K + n]);
  }
  // stage A (64 x F) row-major packed pairs
  if (fullTile) {
    for (int i = tid; i < 64 * FP; i += 256) {
      int r = i / FP, kp = i % FP;
      const float* ap = A + (size_t)(row0 + r) * F + 2 * kp;
      lA[r * AS + kp] = pack2h(ap[0], ap[1]);
    }
  } else {
    for (int i = tid; i < 64 * FP; i += 256) {
      int r = i / FP, kp = i % FP;
      int gr = row0 + r;
      float x0 = 0.f, x1 = 0.f;
      if (gr < M) {
        const float* ap = A + (size_t)gr * F + 2 * kp;
        x0 = ap[0]; x1 = ap[1];
      }
      lA[r * AS + kp] = pack2h(x0, x1);
    }
  }
  __syncthreads();

  const int wave   = tid >> 5;
  const int lane   = tid & 31;
  const int mstrip = wave & 3;           // 16-row strip within tile
  const int chalf  = wave >> 2;          // column half
  const int half   = lane >> 4;          // K-half selector in fragments
  const int lm     = lane & 15;          // M (A frag) / N (B frag, C/D)

  constexpr int TILES = K / 32;          // 16-col tiles per wave
  v8f acc[TILES];
#pragma unroll
  for (int t = 0; t < TILES; ++t) acc[t] = (v8f){};

#pragma unroll
  for (int g = 0; g < F / 32; ++g) {
    union { v16h v; uint4 q[2]; } af;
    const int abase = (mstrip * 16 + lm) * AS + 16 * g + 4 * half;
    af.q[0] = *(const uint4*)&lA[abase];
    af.q[1] = *(const uint4*)&lA[abase + 8];
#pragma unroll
    for (int t = 0; t < TILES; ++t) {
      const int ncol = chalf * (K / 2) + t * 16 + lm;
      union { v16h v; uint4 q[2]; } bf;
      const int bbase = ncol * AS + 16 * g + 4 * half;
      bf.q[0] = *(const uint4*)&lB[bbase];
      bf.q[1] = *(const uint4*)&lB[bbase + 8];
      acc[t] = __builtin_amdgcn_wmma_f32_16x16x32_f16(
          false, af.v, false, bf.v, (short)0, acc[t], false, false);
    }
  }

  // C/D layout: VGPR r -> M = r + 8*half; N = lm
  if (fullTile) {
#pragma unroll
    for (int t = 0; t < TILES; ++t) {
      const int ncol = chalf * (K / 2) + t * 16 + lm;
#pragma unroll
      for (int r = 0; r < 8; ++r) {
        const int grow = row0 + mstrip * 16 + half * 8 + r;
        float v = acc[t][r];
        if (BIAS) v += bias[ncol];
        C[(size_t)grow * K + ncol] = v;
      }
    }
  } else {
#pragma unroll
    for (int t = 0; t < TILES; ++t) {
      const int ncol = chalf * (K / 2) + t * 16 + lm;
#pragma unroll
      for (int r = 0; r < 8; ++r) {
        const int grow = row0 + mstrip * 16 + half * 8 + r;
        if (grow < M) {
          float v = acc[t][r];
          if (BIAS) v += bias[ncol];
          C[(size_t)grow * K + ncol] = v;
        }
      }
    }
  }
}

// ---------------------------------------------------------------------------
// s[n,h] = <h[n,h,:], a_src[h,:]> ; d[n,h] = <h[n,h,:], a_dst[h,:]>
// One wave per node: lane l reads channels l, l+32, ... (coalesced), per-head
// partials reduced across the wave with shfl_xor (wave32).
// ---------------------------------------------------------------------------
template<int H, int C>
__global__ void attn_sd_wave(const float* __restrict__ h, const float* __restrict__ a_src,
                             const float* __restrict__ a_dst,
                             float* __restrict__ s, float* __restrict__ d)
{
  constexpr int HC = H * C;
  constexpr int J  = HC / 32;            // channels per lane
  const int node = blockIdx.x * (blockDim.x >> 5) + (threadIdx.x >> 5);
  const int lane = threadIdx.x & 31;
  if (node >= NNODE) return;             // whole-wave uniform exit
  const float* hp = h + (size_t)node * HC;
  float ps[H], pd[H];
#pragma unroll
  for (int hh = 0; hh < H; ++hh) { ps[hh] = 0.f; pd[hh] = 0.f; }
#pragma unroll
  for (int j = 0; j < J; ++j) {
    const int c  = lane + 32 * j;
    const int hd = (32 * j) / C;         // lane-invariant head index
    const int w  = c - hd * C;           // channel within head
    const float v = hp[c];
    ps[hd] += v * a_src[hd * C + w];
    pd[hd] += v * a_dst[hd * C + w];
  }
#pragma unroll
  for (int hh = 0; hh < H; ++hh) {
    float x = ps[hh], y = pd[hh];
#pragma unroll
    for (int off = 16; off > 0; off >>= 1) {
      x += __shfl_xor(x, off, 32);
      y += __shfl_xor(y, off, 32);
    }
    if (lane == 0) { s[node * H + hh] = x; d[node * H + hh] = y; }
  }
}

__global__ void fill_u32_kernel(unsigned* __restrict__ p, unsigned v, long long n)
{
  long long i = (long long)blockIdx.x * blockDim.x + threadIdx.x;
  if (i < n) p[i] = v;
}

// pass 1: logits + leaky_relu, segment-max via keyed atomicMax
template<int H>
__global__ void attn_edge_pass1(const int* __restrict__ srcIdx, const int* __restrict__ dstIdx,
                                const float* __restrict__ s, const float* __restrict__ d,
                                float* __restrict__ ev, unsigned* __restrict__ mkey)
{
  int e = blockIdx.x * blockDim.x + threadIdx.x;
  if (e >= TOTE) return;
  int src = (e < NNODE) ? e : srcIdx[e - NNODE];
  int dst = (e < NNODE) ? e : dstIdx[e - NNODE];
#pragma unroll
  for (int h = 0; h < H; ++h) {
    float x = s[src * H + h] + d[dst * H + h];
    x = (x >= 0.f) ? x : 0.2f * x;                 // leaky_relu, slope 0.2
    ev[(size_t)e * H + h] = x;
    atomicMax(&mkey[dst * H + h], fkey(x));
  }
}

// pass 2: p = exp(e - m[dst]); segment-sum via atomicAdd
template<int H>
__global__ void attn_edge_pass2(const int* __restrict__ dstIdx,
                                float* __restrict__ ev, const unsigned* __restrict__ mkey,
                                float* __restrict__ zsum)
{
  int e = blockIdx.x * blockDim.x + threadIdx.x;
  if (e >= TOTE) return;
  int dst = (e < NNODE) ? e : dstIdx[e - NNODE];
#pragma unroll
  for (int h = 0; h < H; ++h) {
    float m = funkey(mkey[dst * H + h]);
    float p = __expf(ev[(size_t)e * H + h] - m);
    ev[(size_t)e * H + h] = p;
    atomicAdd(&zsum[dst * H + h], p);
  }
}

// pass 3: out[dst] += alpha * feat[src], float4 granularity
template<int H, int C>
__global__ void attn_edge_aggregate(const int* __restrict__ srcIdx, const int* __restrict__ dstIdx,
                                    const float* __restrict__ p, const float* __restrict__ zsum,
                                    const float* __restrict__ feat, float* __restrict__ out)
{
  constexpr int HC4 = (H * C) / 4;
  long long i = (long long)blockIdx.x * blockDim.x + threadIdx.x;
  if (i >= (long long)TOTE * HC4) return;
  int e = (int)(i / HC4);
  int q = (int)(i % HC4);                          // float4 chunk index
  int h = (q * 4) / C;
  int src = (e < NNODE) ? e : srcIdx[e - NNODE];
  int dst = (e < NNODE) ? e : dstIdx[e - NNODE];
  float alpha = p[(size_t)e * H + h] / (zsum[dst * H + h] + 1e-16f);
  float4 f = *(const float4*)(feat + (size_t)src * (H * C) + q * 4);
  float* o = out + (size_t)dst * (H * C) + q * 4;
  atomicAdd(o + 0, alpha * f.x);
  atomicAdd(o + 1, alpha * f.y);
  atomicAdd(o + 2, alpha * f.z);
  atomicAdd(o + 3, alpha * f.w);
}

template<bool ELU>
__global__ void bias_act(float* __restrict__ x, const float* __restrict__ b, int k)
{
  long long i = (long long)blockIdx.x * blockDim.x + threadIdx.x;
  if (i >= (long long)NNODE * k) return;
  float v = x[i] + b[(int)(i % k)];
  if (ELU) v = (v > 0.f) ? v : (__expf(v) - 1.f);
  x[i] = v;
}

__global__ void edge_dot_kernel(const int* __restrict__ prov, const int* __restrict__ memb,
                                const float* __restrict__ zp, const float* __restrict__ zm,
                                float* __restrict__ out)
{
  int i = blockIdx.x * blockDim.x + threadIdx.x;
  if (i >= NEDGE) return;
  const float4* a = (const float4*)(zp + (size_t)prov[i] * 64);
  const float4* b = (const float4*)(zm + (size_t)memb[i] * 64);
  float sum = 0.f;
#pragma unroll
  for (int k = 0; k < 16; ++k) {
    float4 x = a[k], y = b[k];
    sum += x.x * y.x + x.y * y.y + x.z * y.z + x.w * y.w;
  }
  out[i] = sum;
}

// ---------------------------------------------------------------------------
#define CEIL(a, b) (int)(((long long)(a) + (b) - 1) / (b))

static void run_side(const float* x, const int* srcIdx, const int* dstIdx,
                     const float* W1, const float* as1, const float* ad1, const float* b1,
                     const float* W2, const float* as2, const float* ad2, const float* b2,
                     const float* Wd, const float* bd,
                     float* H1, float* XM, float* Z, float* SB, float* DB,
                     unsigned* MK, float* ZS, float* PB,
                     float* xhat, hipStream_t stream)
{
  const int GG = CEIL(NNODE, 64);
  const int WG = CEIL(NNODE, 8);   // wave-per-node kernels, 8 waves/block
  // ---- layer 1: heads=4, ch=32 ----
  gemm_wmma<128, 128, false><<<GG, 256, 0, stream>>>(x, W1, nullptr, H1, NNODE);
  attn_sd_wave<4, 32><<<WG, 256, 0, stream>>>(H1, as1, ad1, SB, DB);
  fill_u32_kernel<<<CEIL(NNODE * 4, 256), 256, 0, stream>>>(MK, NEG_INF_KEY, NNODE * 4);
  fill_u32_kernel<<<CEIL(NNODE * 4, 256), 256, 0, stream>>>((unsigned*)ZS, 0u, NNODE * 4);
  fill_u32_kernel<<<CEIL((long long)NNODE * 128, 256), 256, 0, stream>>>((unsigned*)XM, 0u, (long long)NNODE * 128);
  attn_edge_pass1<4><<<CEIL(TOTE, 256), 256, 0, stream>>>(srcIdx, dstIdx, SB, DB, PB, MK);
  attn_edge_pass2<4><<<CEIL(TOTE, 256), 256, 0, stream>>>(dstIdx, PB, MK, ZS);
  attn_edge_aggregate<4, 32><<<CEIL((long long)TOTE * 32, 256), 256, 0, stream>>>(srcIdx, dstIdx, PB, ZS, H1, XM);
  bias_act<true><<<CEIL((long long)NNODE * 128, 256), 256, 0, stream>>>(XM, b1, 128);
  // ---- layer 2: heads=1, ch=64 (H1 reused as h2 buffer, N*64) ----
  gemm_wmma<128, 64, false><<<GG, 256, 0, stream>>>(XM, W2, nullptr, H1, NNODE);
  attn_sd_wave<1, 64><<<WG, 256, 0, stream>>>(H1, as2, ad2, SB, DB);
  fill_u32_kernel<<<CEIL(NNODE, 256), 256, 0, stream>>>(MK, NEG_INF_KEY, NNODE);
  fill_u32_kernel<<<CEIL(NNODE, 256), 256, 0, stream>>>((unsigned*)ZS, 0u, NNODE);
  fill_u32_kernel<<<CEIL((long long)NNODE * 64, 256), 256, 0, stream>>>((unsigned*)Z, 0u, (long long)NNODE * 64);
  attn_edge_pass1<1><<<CEIL(TOTE, 256), 256, 0, stream>>>(srcIdx, dstIdx, SB, DB, PB, MK);
  attn_edge_pass2<1><<<CEIL(TOTE, 256), 256, 0, stream>>>(dstIdx, PB, MK, ZS);
  attn_edge_aggregate<1, 64><<<CEIL((long long)TOTE * 16, 256), 256, 0, stream>>>(srcIdx, dstIdx, PB, ZS, H1, Z);
  bias_act<false><<<CEIL((long long)NNODE * 64, 256), 256, 0, stream>>>(Z, b2, 64);
  // ---- decoder: z (N x 64) @ Wd (64 x 128) + bd ----
  gemm_wmma<64, 128, true><<<GG, 256, 0, stream>>>(Z, Wd, bd, xhat, NNODE);
}

extern "C" void kernel_launch(void* const* d_in, const int* in_sizes, int n_in,
                              void* d_out, int out_size, void* d_ws, size_t ws_size,
                              hipStream_t stream)
{
  const float* x_m  = (const float*)d_in[0];
  const float* x_p  = (const float*)d_in[1];
  const int*   edge = (const int*)d_in[2];
  const int*   prov = edge;           // edge_index[0]
  const int*   memb = edge + NEDGE;   // edge_index[1]
  const float* W1m = (const float*)d_in[3];
  const float* as1m = (const float*)d_in[4];
  const float* ad1m = (const float*)d_in[5];
  const float* b1m = (const float*)d_in[6];
  const float* W2m = (const float*)d_in[7];
  const float* as2m = (const float*)d_in[8];
  const float* ad2m = (const float*)d_in[9];
  const float* b2m = (const float*)d_in[10];
  const float* W1p = (const float*)d_in[11];
  const float* as1p = (const float*)d_in[12];
  const float* ad1p = (const float*)d_in[13];
  const float* b1p = (const float*)d_in[14];
  const float* W2p = (const float*)d_in[15];
  const float* as2p = (const float*)d_in[16];
  const float* ad2p = (const float*)d_in[17];
  const float* b2p = (const float*)d_in[18];
  const float* Wdm = (const float*)d_in[19];
  const float* bdm = (const float*)d_in[20];
  const float* Wdp = (const float*)d_in[21];
  const float* bdp = (const float*)d_in[22];

  // workspace layout (floats)
  float* ws = (float*)d_ws;
  float* H1 = ws;                                   // N*128 (reused as h2: N*64)
  float* XM = H1 + (size_t)NNODE * 128;             // N*128 (layer-1 output, both sides)
  float* ZM = XM + (size_t)NNODE * 128;             // N*64
  float* ZP = ZM + (size_t)NNODE * 64;              // N*64
  float* SB = ZP + (size_t)NNODE * 64;              // N*4
  float* DB = SB + (size_t)NNODE * 4;               // N*4
  unsigned* MK = (unsigned*)(DB + (size_t)NNODE * 4); // N*4
  float* ZS = (float*)(MK + (size_t)NNODE * 4);     // N*4
  float* PB = ZS + (size_t)NNODE * 4;               // TOTE*4

  float* out_xm = (float*)d_out;
  float* out_xp = out_xm + (size_t)NNODE * 128;
  float* out_el = out_xp + (size_t)NNODE * 128;

  // member graph: src = provider, dst = member
  run_side(x_m, prov, memb, W1m, as1m, ad1m, b1m, W2m, as2m, ad2m, b2m, Wdm, bdm,
           H1, XM, ZM, SB, DB, MK, ZS, PB, out_xm, stream);
  // provider graph: src = member, dst = provider
  run_side(x_p, memb, prov, W1p, as1p, ad1p, b1p, W2p, as2p, ad2p, b2p, Wdp, bdp,
           H1, XM, ZP, SB, DB, MK, ZS, PB, out_xp, stream);

  edge_dot_kernel<<<CEIL(NEDGE, 256), 256, 0, stream>>>(prov, memb, ZP, ZM, out_el);
}
